// BlockAttnResLayer_64871186039061
// MI455X (gfx1250) — matrix operand order repaired and checked by reading, not compiled, not run on hardware
//
#include <hip/hip_runtime.h>
#include <hip/hip_bf16.h>

// ---------------------------------------------------------------------------
// Problem constants (fixed by setup_inputs)
// ---------------------------------------------------------------------------
static constexpr int NB  = 8;       // completed blocks
static constexpr int N1  = 9;       // blocks + partial
static constexpr int BB  = 2;       // batch
static constexpr int TT  = 2048;    // tokens
static constexpr int DD  = 2048;    // hidden
static constexpr int FF  = 8192;    // ffn hidden
static constexpr int MM  = BB * TT; // 4096 rows

typedef __attribute__((ext_vector_type(16))) __bf16 v16bf;
typedef __attribute__((ext_vector_type(8)))  float  v8f;

union Frag { int4 i4[2]; v16bf v; };

__device__ __forceinline__ unsigned short f2bf(float f) {
    unsigned int u = __float_as_uint(f);
    // round-to-nearest-even bf16
    u += 0x7FFFu + ((u >> 16) & 1u);
    return (unsigned short)(u >> 16);
}

__device__ __forceinline__ float gelu_tanh(float x) {
    float x3 = x * x * x;
    return 0.5f * x * (1.0f + tanhf(0.7978845608028654f * (x + 0.044715f * x3)));
}

// CDNA5 async DMA: global -> LDS, 16 bytes per lane, tracked by ASYNCcnt.
// ldsOff: LDS byte address (low 32 bits of a generic pointer to __shared__,
// per ISA 10.2).
__device__ __forceinline__ void async_copy_b128(unsigned ldsOff, const void* gptr) {
    asm volatile("global_load_async_to_lds_b128 %0, %1, off"
                 :: "v"(ldsOff), "v"(gptr)
                 : "memory");
}

__device__ __forceinline__ void wait_async0() {
    asm volatile("s_wait_asynccnt 0x0" ::: "memory");
}

// ---------------------------------------------------------------------------
// Kernel 1: fused RMSNorm-logit / block-softmax / weighted sum
// One workgroup per (b,t). 256 threads x 8 elements = D=2048.
// V slice kept in registers: HBM read exactly once.
// ---------------------------------------------------------------------------
__global__ __launch_bounds__(256)
void attn_pool_kernel(const float* __restrict__ blocks,
                      const float* __restrict__ partial,
                      const float* __restrict__ proj_w,
                      const float* __restrict__ norm_scale,
                      float* __restrict__ h_out,
                      unsigned short* __restrict__ x_bf16)
{
    __shared__ float red_ssq[N1][8];
    __shared__ float red_dot[N1][8];
    __shared__ float alpha_sh[N1];

    const int bt  = blockIdx.x;           // 0..B*T-1
    const int b   = bt / TT;
    const int t   = bt % TT;
    const int tid = threadIdx.x;
    const int d0  = tid * 8;

    // fused projection weight w[d] = norm_scale[d]*proj_w[d]
    float4 ns0 = *(const float4*)(norm_scale + d0);
    float4 ns1 = *(const float4*)(norm_scale + d0 + 4);
    float4 pw0 = *(const float4*)(proj_w + d0);
    float4 pw1 = *(const float4*)(proj_w + d0 + 4);
    float w[8] = { ns0.x*pw0.x, ns0.y*pw0.y, ns0.z*pw0.z, ns0.w*pw0.w,
                   ns1.x*pw1.x, ns1.y*pw1.y, ns1.z*pw1.z, ns1.w*pw1.w };

    float v[N1][8];
    float ssq[N1], dt[N1];

#pragma unroll
    for (int n = 0; n < N1; ++n) {
        const float* base = (n < NB)
            ? blocks  + (((size_t)n * BB + b) * TT + t) * DD
            : partial + ((size_t)b * TT + t) * DD;
        float4 a = *(const float4*)(base + d0);
        float4 c = *(const float4*)(base + d0 + 4);
        v[n][0]=a.x; v[n][1]=a.y; v[n][2]=a.z; v[n][3]=a.w;
        v[n][4]=c.x; v[n][5]=c.y; v[n][6]=c.z; v[n][7]=c.w;
        float s = 0.f, d = 0.f;
#pragma unroll
        for (int i = 0; i < 8; ++i) { s += v[n][i]*v[n][i]; d += w[i]*v[n][i]; }
        ssq[n] = s; dt[n] = d;
    }

    // wave32 butterfly reduce
#pragma unroll
    for (int off = 16; off > 0; off >>= 1) {
#pragma unroll
        for (int n = 0; n < N1; ++n) {
            ssq[n] += __shfl_xor(ssq[n], off, 32);
            dt[n]  += __shfl_xor(dt[n],  off, 32);
        }
    }
    const int wave = tid >> 5;
    if ((tid & 31) == 0) {
#pragma unroll
        for (int n = 0; n < N1; ++n) { red_ssq[n][wave] = ssq[n]; red_dot[n][wave] = dt[n]; }
    }
    __syncthreads();

    if (tid == 0) {
        float logit[N1];
#pragma unroll
        for (int n = 0; n < N1; ++n) {
            float s = 0.f, d = 0.f;
#pragma unroll
            for (int wv = 0; wv < 8; ++wv) { s += red_ssq[n][wv]; d += red_dot[n][wv]; }
            float rms = sqrtf(s * (1.0f / DD) + 1e-8f);
            logit[n] = d / rms;
        }
        float mx = logit[0];
#pragma unroll
        for (int n = 1; n < N1; ++n) mx = fmaxf(mx, logit[n]);
        float ssum = 0.f;
        float e[N1];
#pragma unroll
        for (int n = 0; n < N1; ++n) { e[n] = __expf(logit[n] - mx); ssum += e[n]; }
        float inv = 1.0f / ssum;
#pragma unroll
        for (int n = 0; n < N1; ++n) alpha_sh[n] = e[n] * inv;
    }
    __syncthreads();

    float alpha[N1];
#pragma unroll
    for (int n = 0; n < N1; ++n) alpha[n] = alpha_sh[n];

    float hv[8];
#pragma unroll
    for (int i = 0; i < 8; ++i) {
        float acc = 0.f;
#pragma unroll
        for (int n = 0; n < N1; ++n) acc += alpha[n] * v[n][i];
        hv[i] = acc;
    }

    // h (f32) -> d_out, X (bf16) -> workspace
    float* hp = h_out + (size_t)bt * DD + d0;
    float4 o0 = { hv[0], hv[1], hv[2], hv[3] };
    float4 o1 = { hv[4], hv[5], hv[6], hv[7] };
    *(float4*)(hp)     = o0;
    *(float4*)(hp + 4) = o1;

    unsigned int pk[4];
#pragma unroll
    for (int i = 0; i < 4; ++i)
        pk[i] = (unsigned int)f2bf(hv[2*i]) | ((unsigned int)f2bf(hv[2*i+1]) << 16);
    int4 xo = { (int)pk[0], (int)pk[1], (int)pk[2], (int)pk[3] };
    *(int4*)(x_bf16 + (size_t)bt * DD + d0) = xo;
}

// ---------------------------------------------------------------------------
// Kernel 2: f32 [R][C] -> bf16 [C][R]  (transpose + downconvert, LDS-tiled)
// ---------------------------------------------------------------------------
__global__ __launch_bounds__(256)
void transpose_to_bf16_kernel(const float* __restrict__ in,
                              unsigned short* __restrict__ out,
                              int R, int C)
{
    __shared__ float tile[32][33];
    const int cb = blockIdx.x * 32;
    const int rb = blockIdx.y * 32;
    const int tx = threadIdx.x;   // 0..31
    const int ty = threadIdx.y;   // 0..7
#pragma unroll
    for (int j = 0; j < 4; ++j)
        tile[ty + 8*j][tx] = in[(size_t)(rb + ty + 8*j) * C + cb + tx];
    __syncthreads();
#pragma unroll
    for (int j = 0; j < 4; ++j)
        out[(size_t)(cb + ty + 8*j) * R + rb + tx] = f2bf(tile[tx][ty + 8*j]);
}

// ---------------------------------------------------------------------------
// Kernel 3/4: bf16 WMMA GEMM.  C[m][n] = sum_k A[m][k] * Bt[n][k]
//   A : [M][K] bf16 row-major, Bt : [N][K] bf16 (i.e. B transposed)
//   128x128 tile per WG, BK=32, 8 waves -> each wave 64x32 = 8 WMMA accs.
//   Double-buffered LDS filled by GLOBAL_LOAD_ASYNC_TO_LDS_B128 (ASYNCcnt),
//   one barrier per K-step; next tile DMA overlaps current WMMA work.
//   GELU=true : outBf[m][n] = bf16(gelu(c))
//   GELU=false: outF[m][n]  = residual[m][n] + c
// ---------------------------------------------------------------------------
template <bool GELU>
__global__ __launch_bounds__(256)
void gemm_bf16_wmma_kernel(const unsigned short* __restrict__ A,
                           const unsigned short* __restrict__ Bt,
                           int M, int N, int K,
                           unsigned short* __restrict__ outBf,
                           float* __restrict__ outF,
                           const float* __restrict__ residual)
{
    __shared__ unsigned short As[2][128][40];   // +8 pad (80B row stride), ping-pong
    __shared__ unsigned short Bs[2][128][40];

    const int tid  = threadIdx.x;
    const int bn   = blockIdx.x * 128;
    const int bm   = blockIdx.y * 128;

    const int wave  = tid >> 5;
    const int lane  = tid & 31;
    const int wm    = (wave & 1) * 64;    // 2 waves along M
    const int wn    = (wave >> 1) * 32;   // 4 waves along N
    const int lrow  = lane & 15;
    const int lhalf = lane >> 4;

    const int ldRow = tid >> 1;           // 0..127
    const int ldSeg = (tid & 1) * 16;     // 0 / 16 elements (32B each)

    v8f acc[4][2];
    const v8f vz = { 0.f,0.f,0.f,0.f,0.f,0.f,0.f,0.f };
#pragma unroll
    for (int mi = 0; mi < 4; ++mi)
#pragma unroll
        for (int ni = 0; ni < 2; ++ni) acc[mi][ni] = vz;

    const unsigned short* aPtr = A  + (size_t)(bm + ldRow) * K + ldSeg;
    const unsigned short* bPtr = Bt + (size_t)(bn + ldRow) * K + ldSeg;

    // LDS byte offsets for this thread's DMA destinations (low 32 bits of
    // the generic pointer == LDS offset).
    const unsigned bufStride = 128u * 40u * 2u;   // 10240 B per matrix buffer
    const unsigned ldsA0 = (unsigned)(uintptr_t)&As[0][ldRow][ldSeg];
    const unsigned ldsB0 = (unsigned)(uintptr_t)&Bs[0][ldRow][ldSeg];

    auto issue_tile = [&](int k0, int buf) {
        const unsigned la = ldsA0 + (unsigned)buf * bufStride;
        const unsigned lb = ldsB0 + (unsigned)buf * bufStride;
        async_copy_b128(la,      aPtr + k0);
        async_copy_b128(la + 16, aPtr + k0 + 8);
        async_copy_b128(lb,      bPtr + k0);
        async_copy_b128(lb + 16, bPtr + k0 + 8);
    };

    issue_tile(0, 0);                       // prologue DMA

    int buf = 0;
    for (int k0 = 0; k0 < K; k0 += 32) {
        wait_async0();                      // my DMA for `buf` landed
        __syncthreads();                    // everyone's landed; prev reads of buf^1 done
        if (k0 + 32 < K)
            issue_tile(k0 + 32, buf ^ 1);   // overlap next tile DMA with WMMA

        // A frag: lane=row (M), VGPR0-3: K = lhalf*8..+7, VGPR4-7: K = 16+lhalf*8..+7
        Frag afr[4];
#pragma unroll
        for (int mi = 0; mi < 4; ++mi) {
            const unsigned short* p = &As[buf][wm + mi*16 + lrow][lhalf * 8];
            afr[mi].i4[0] = *(const int4*)(p);
            afr[mi].i4[1] = *(const int4*)(p + 16);
        }
        // B frag: lane=col (N), VGPR0-3: K = lhalf*8..+7, VGPR4-7: K = 16+lhalf*8..+7
        Frag bfr[2];
#pragma unroll
        for (int ni = 0; ni < 2; ++ni) {
            const unsigned short* p = &Bs[buf][wn + ni*16 + lrow][lhalf * 8];
            bfr[ni].i4[0] = *(const int4*)(p);
            bfr[ni].i4[1] = *(const int4*)(p + 16);
        }
#pragma unroll
        for (int mi = 0; mi < 4; ++mi)
#pragma unroll
            for (int ni = 0; ni < 2; ++ni)
                acc[mi][ni] = __builtin_amdgcn_wmma_f32_16x16x32_bf16(
                    false, afr[mi].v, false, bfr[ni].v,
                    (short)0, acc[mi][ni], false, false);

        buf ^= 1;
    }

    // C/D layout: lane L holds col N=L%16; VGPR r holds row M = r + 8*(L/16)
#pragma unroll
    for (int mi = 0; mi < 4; ++mi) {
#pragma unroll
        for (int ni = 0; ni < 2; ++ni) {
#pragma unroll
            for (int r = 0; r < 8; ++r) {
                int row = bm + wm + mi*16 + lhalf*8 + r;
                int col = bn + wn + ni*16 + lrow;
                size_t idx = (size_t)row * N + col;
                float val = acc[mi][ni][r];
                if constexpr (GELU) {
                    outBf[idx] = f2bf(gelu_tanh(val));
                } else {
                    outF[idx] = residual[idx] + val;
                }
            }
        }
    }
}

// ---------------------------------------------------------------------------
// Launch
// ---------------------------------------------------------------------------
extern "C" void kernel_launch(void* const* d_in, const int* in_sizes, int n_in,
                              void* d_out, int out_size, void* d_ws, size_t ws_size,
                              hipStream_t stream) {
    const float* blocks     = (const float*)d_in[0];
    const float* partial    = (const float*)d_in[1];
    const float* proj_w     = (const float*)d_in[2];
    const float* norm_scale = (const float*)d_in[3];
    const float* ffn_w1     = (const float*)d_in[4];
    const float* ffn_w2     = (const float*)d_in[5];

    float* h_out       = (float*)d_out;                       // [M][D]
    float* new_partial = (float*)d_out + (size_t)MM * DD;     // [M][D]

    // workspace layout (bytes)
    constexpr size_t X_BYTES   = (size_t)MM * DD * 2;   // 16 MiB
    constexpr size_t W1T_BYTES = (size_t)FF * DD * 2;   // 32 MiB
    constexpr size_t W2T_BYTES = (size_t)DD * FF * 2;   // 32 MiB
    char* ws = (char*)d_ws;
    unsigned short* Xbf = (unsigned short*)(ws);
    unsigned short* W1t = (unsigned short*)(ws + X_BYTES);
    unsigned short* W2t = (unsigned short*)(ws + X_BYTES + W1T_BYTES);
    unsigned short* Act = (unsigned short*)(ws + X_BYTES + W1T_BYTES + W2T_BYTES);

    // 1) block softmax pooling -> h (f32) + X (bf16)
    attn_pool_kernel<<<MM, 256, 0, stream>>>(blocks, partial, proj_w, norm_scale,
                                             h_out, Xbf);

    // 2) weight downconvert + transpose to [N][K]
    transpose_to_bf16_kernel<<<dim3(FF/32, DD/32), dim3(32, 8), 0, stream>>>(
        ffn_w1, W1t, DD, FF);
    transpose_to_bf16_kernel<<<dim3(DD/32, FF/32), dim3(32, 8), 0, stream>>>(
        ffn_w2, W2t, FF, DD);

    // 3) Act = gelu(X @ W1)   (M=4096, N=8192, K=2048)
    gemm_bf16_wmma_kernel<true><<<dim3(FF/128, MM/128), 256, 0, stream>>>(
        Xbf, W1t, MM, FF, DD, Act, nullptr, nullptr);

    // 4) new_partial = partial + Act @ W2   (M=4096, N=2048, K=8192)
    gemm_bf16_wmma_kernel<false><<<dim3(DD/128, MM/128), 256, 0, stream>>>(
        Act, W2t, MM, DD, FF, nullptr, new_partial, partial);
}
